// Decoder_7069516169250
// MI455X (gfx1250) — matrix operand order repaired and checked by reading, not compiled, not run on hardware
//
#include <hip/hip_runtime.h>
#include <hip/hip_bf16.h>
#include <stdint.h>

// ---------------------------------------------------------------------------
// Types for CDNA5 WMMA (wave32). bf16 16x16x32, f32 accumulate.
// ---------------------------------------------------------------------------
typedef __attribute__((ext_vector_type(16))) __bf16    v16bf;
typedef __attribute__((ext_vector_type(8)))  float     v8f;
typedef __attribute__((ext_vector_type(4)))  unsigned  u32x4;

union FragBF { u32x4 q[2]; v16bf v; };

// Native converts (v_cvt_*_bf16) instead of manual bit rounding.
static __device__ __forceinline__ __bf16 f2bf(float f) { return (__bf16)f; }

// Problem constants
#define BD   32       // batch
#define LX   64       // encoder length
#define TT   64       // decoder steps
#define DM   1024     // d_model
#define VV   32000    // vocab

// ---------------------------------------------------------------------------
// Fragment helpers.
// A frag (16x32): lane ln=L&15 is row M, hf=L>>4; runs K = kb+8h+[0,8) and
// kb+16+8h+[0,8). B frag (32x16): lane ln is col N, same K runs as A.
// ---------------------------------------------------------------------------
static __device__ __forceinline__ void load_a_bf(const __bf16* aRow, int kb, FragBF& a) {
    a.q[0] = *(const u32x4*)(aRow + kb);
    a.q[1] = *(const u32x4*)(aRow + kb + 16);
}
static __device__ __forceinline__ void load_a_f32(const float* r, int kb, FragBF& a) {
#pragma unroll
    for (int e = 0; e < 8; ++e) a.v[e] = f2bf(r[kb + e]);
#pragma unroll
    for (int e = 0; e < 8; ++e) a.v[8 + e] = f2bf(r[kb + 16 + e]);
}
static __device__ __forceinline__ void load_b4(const __bf16* bBase, size_t ldb, int kb,
                                               FragBF (&b)[4]) {
#pragma unroll
    for (int nt = 0; nt < 4; ++nt) {
        const __bf16* bp = bBase + (size_t)nt * 16 * ldb + kb;
        b[nt].q[0] = *(const u32x4*)bp;
        b[nt].q[1] = *(const u32x4*)(bp + 16);
    }
}
static __device__ __forceinline__ void wmma4(const FragBF& a, const FragBF (&b)[4],
                                             v8f (&acc)[4]) {
#pragma unroll
    for (int nt = 0; nt < 4; ++nt)
        acc[nt] = __builtin_amdgcn_wmma_f32_16x16x32_bf16(
            false, a.v, false, b[nt].v, (short)0, acc[nt], false, false);
}

// CDNA5 async global->LDS copy (ASYNCcnt-tracked; no VGPR round trip).
// LDS flat pointers keep the allocation offset in addr[31:0].
static __device__ __forceinline__ void async_b128(const __bf16* lds_dst, const __bf16* gsrc) {
    unsigned      l = (unsigned)(size_t)lds_dst;
    unsigned long long g = (unsigned long long)(size_t)gsrc;
    asm volatile("global_load_async_to_lds_b128 %0, %1, off" :: "v"(l), "v"(g) : "memory");
}

// ---------------------------------------------------------------------------
// Prep kernels (run once per launch)
// ---------------------------------------------------------------------------
__global__ __launch_bounds__(256) void k_prep_wfull(
    const float* __restrict__ Wih,  // [4096,3072]
    const float* __restrict__ Whh,  // [4096,1024]
    __bf16* __restrict__ wf)        // [4096,4096]  row n = [Wih[n,:], Whh[n,:]]
{
    size_t id = (size_t)blockIdx.x * blockDim.x + threadIdx.x;
    if (id >= (size_t)4096 * 4096) return;
    int n = (int)(id >> 12), k = (int)(id & 4095);
    float v = (k < 3072) ? Wih[(size_t)n * 3072 + k]
                         : Whh[(size_t)n * 1024 + (k - 3072)];
    wf[id] = f2bf(v);
}

__global__ __launch_bounds__(256) void k_f2bf(
    const float* __restrict__ src, __bf16* __restrict__ dst, int n)
{
    int id = blockIdx.x * blockDim.x + threadIdx.x;
    if (id < n) dst[id] = f2bf(src[id]);
}

__global__ __launch_bounds__(256) void k_bsum(
    const float* __restrict__ a, const float* __restrict__ b,
    float* __restrict__ o, int n)
{
    int id = blockIdx.x * blockDim.x + threadIdx.x;
    if (id < n) o[id] = a[id] + b[id];
}

// c <- c0 ; h into xfh[:,2048:] and xh[:,0:1024] ; input-feed region <- 0
__global__ __launch_bounds__(256) void k_init(
    const float* __restrict__ h0, const float* __restrict__ c0,
    float* __restrict__ c, __bf16* __restrict__ xfh, __bf16* __restrict__ xh)
{
    int id = blockIdx.x * blockDim.x + threadIdx.x;   // 32*2048
    if (id >= BD * 2048) return;
    int b = id >> 11, j = id & 2047;
    if (j < DM) {
        c[b * DM + j] = c0[b * DM + j];
        __bf16 hb = f2bf(h0[b * DM + j]);
        xfh[(size_t)b * 3072 + 2048 + j] = hb;   // h region
        xh[(size_t)b * 3072 + j]         = hb;
    }
    xfh[(size_t)b * 3072 + j] = f2bf(0.0f);      // input_feed = 0
}

// ---------------------------------------------------------------------------
// Gate GEMM with fused embedding gather:
//   gates[32,4096] = [emb|feed|h] @ [W_ih|W_hh]^T + (b_ih+b_hh)
// Phase 1 (K 0..1024): A rows gathered from fp32 word_emb[y[b,t]] + cvt.
// Phase 2 (K 1024..4096): A from bf16 xfh (feed at col0, h at col 2048).
// Double-buffered fragment pipeline; 128 waves (2 m-tiles x 64 n-groups).
// ---------------------------------------------------------------------------
__global__ __launch_bounds__(256) void k_gates(
    const float* __restrict__ word_emb, const int* __restrict__ y, int t,
    const __bf16* __restrict__ xfh, const __bf16* __restrict__ wfull,
    const float* __restrict__ bsum, float* __restrict__ gates)
{
    int wid  = (blockIdx.x * blockDim.x + threadIdx.x) >> 5;
    int lane = threadIdx.x & 31;
    if (wid >= 128) return;
    int mt = wid & 1, ng = wid >> 1;
    int nBase = ng * 64, mBase = mt * 16;
    int ln = lane & 15, hf = lane >> 4;
    int row = mBase + ln;                         // batch index

    int tok = y[row * TT + t];
    const float*  eRow  = word_emb + (size_t)tok * DM + 8 * hf;
    const __bf16* bBase = wfull + (size_t)(nBase + ln) * 4096 + 8 * hf;

    v8f acc[4] = {};
    FragBF aC, aN, bC[4], bN[4];

    // ---- phase 1: K in [0,1024), fp32 embedding gather ----
    load_a_f32(eRow, 0, aC); load_b4(bBase, 4096, 0, bC);
    for (int kb = 0; kb < 1024; kb += 64) {
        load_a_f32(eRow, kb + 32, aN); load_b4(bBase, 4096, kb + 32, bN);
        wmma4(aC, bC, acc);
        if (kb + 64 < 1024) { load_a_f32(eRow, kb + 64, aC); load_b4(bBase, 4096, kb + 64, bC); }
        wmma4(aN, bN, acc);
    }
    // ---- phase 2: K in [1024,4096), bf16 [feed|h] ----
    const __bf16* aRow = xfh + (size_t)row * 3072 + 8 * hf - 1024;
    load_a_bf(aRow, 1024, aC); load_b4(bBase, 4096, 1024, bC);
    for (int kb = 1024; kb < 4096; kb += 64) {
        load_a_bf(aRow, kb + 32, aN); load_b4(bBase, 4096, kb + 32, bN);
        wmma4(aC, bC, acc);
        if (kb + 64 < 4096) { load_a_bf(aRow, kb + 64, aC); load_b4(bBase, 4096, kb + 64, bC); }
        wmma4(aN, bN, acc);
    }

#pragma unroll
    for (int nt = 0; nt < 4; ++nt) {
        int col = nBase + nt * 16 + ln;
        float bv = bsum[col];
#pragma unroll
        for (int r = 0; r < 8; ++r) {
            int orow = mBase + 8 * hf + r;
            gates[(size_t)orow * 4096 + col] = acc[nt][r] + bv;
        }
    }
}

// ---------------------------------------------------------------------------
// Generic M=32 bf16 WMMA GEMM (K multiple of 64):
//   out[32,N] = A[32,K](lda) x W[N,K](ldb)^T (+bias)(tanh?)
// One wave = 16x64 tile, double-buffered fragments.
// ---------------------------------------------------------------------------
__global__ __launch_bounds__(256) void k_gemm32(
    const __bf16* __restrict__ A, int lda,
    const __bf16* __restrict__ Wt, int ldb,
    const float* __restrict__ bias,
    float* __restrict__ outF, __bf16* __restrict__ outB,
    int N, int K, int doTanh)
{
    int wid  = (blockIdx.x * blockDim.x + threadIdx.x) >> 5;
    int lane = threadIdx.x & 31;
    int totalWaves = 2 * (N >> 6);
    if (wid >= totalWaves) return;

    int mt = wid & 1, ng = wid >> 1;
    int nBase = ng * 64, mBase = mt * 16;
    int ln = lane & 15, hf = lane >> 4;

    const __bf16* aRow  = A  + (size_t)(mBase + ln) * lda + 8 * hf;
    const __bf16* bBase = Wt + (size_t)(nBase + ln) * ldb + 8 * hf;

    v8f acc[4] = {};
    FragBF aC, aN, bC[4], bN[4];
    load_a_bf(aRow, 0, aC); load_b4(bBase, (size_t)ldb, 0, bC);
    for (int kb = 0; kb < K; kb += 64) {
        load_a_bf(aRow, kb + 32, aN); load_b4(bBase, (size_t)ldb, kb + 32, bN);
        wmma4(aC, bC, acc);
        if (kb + 64 < K) { load_a_bf(aRow, kb + 64, aC); load_b4(bBase, (size_t)ldb, kb + 64, bC); }
        wmma4(aN, bN, acc);
    }

#pragma unroll
    for (int nt = 0; nt < 4; ++nt) {
        int col = nBase + nt * 16 + ln;
        float bv = bias ? bias[col] : 0.0f;
#pragma unroll
        for (int r = 0; r < 8; ++r) {
            int row = mBase + 8 * hf + r;
            float v = acc[nt][r] + bv;
            if (doTanh) v = tanhf(v);
            if (outF) outF[(size_t)row * N + col] = v;
            if (outB) outB[(size_t)row * N + col] = f2bf(v);
        }
    }
}

// ---------------------------------------------------------------------------
// LSTM pointwise: gates [32,4096] (i,f,g,o), in-place c update, h -> bf16
// into xfh h-region (recurrence) and xh[:,0:1024] (pre-readout A).
// ---------------------------------------------------------------------------
__global__ __launch_bounds__(256) void k_lstm(
    const float* __restrict__ gates, float* __restrict__ c,
    __bf16* __restrict__ xfh, __bf16* __restrict__ xh)
{
    int id = blockIdx.x * blockDim.x + threadIdx.x;   // 32*1024
    if (id >= BD * DM) return;
    int b = id >> 10, d = id & 1023;
    const float* g = gates + (size_t)b * 4096;
    float ig = g[d], fg = g[1024 + d], gg = g[2048 + d], og = g[3072 + d];
    float si = 1.0f / (1.0f + __expf(-ig));
    float sf = 1.0f / (1.0f + __expf(-fg));
    float so = 1.0f / (1.0f + __expf(-og));
    float cn = sf * c[id] + si * tanhf(gg);
    float hn = so * tanhf(cn);
    c[id] = cn;
    __bf16 hb = f2bf(hn);
    xfh[(size_t)b * 3072 + 2048 + d] = hb;
    xh[(size_t)b * 3072 + d]         = hb;
}

// ---------------------------------------------------------------------------
// Fused attention: scores (additive, fp32 tanh, wave reduce) -> softmax ->
// context.  ctx is written as bf16 into BOTH next-step input-feed
// (xfh[:,0:2048]) and the pre-readout A matrix (xh[:,1024:3072]).
// One block per batch row, 8 waves.
// ---------------------------------------------------------------------------
__global__ __launch_bounds__(256) void k_att(
    const float* __restrict__ x_enc_k, const float* __restrict__ x_enc,
    const float* __restrict__ tmp, const float* __restrict__ w_att,
    const float* __restrict__ w_att_b, const unsigned char* __restrict__ mask,
    __bf16* __restrict__ xfh, __bf16* __restrict__ xh)
{
    __shared__ float p[LX];
    int b = blockIdx.x, tid = threadIdx.x;
    int lane = tid & 31, w = tid >> 5;
    const float* tp = tmp + (size_t)b * DM;

#pragma unroll 2
    for (int i = 0; i < 8; ++i) {                 // wave w -> rows l = w*8+i
        int l = w * 8 + i;
        const float* xk = x_enc_k + ((size_t)b * LX + l) * DM;
        float s = 0.0f;
        for (int d = lane; d < DM; d += 32)
            s += tanhf(xk[d] + tp[d]) * w_att[d];
#pragma unroll
        for (int off = 16; off; off >>= 1) s += __shfl_xor(s, off, 32);
        if (lane == 0) {
            float v = s + w_att_b[0];
            if (mask[b * LX + l]) v = -1e9f;
            p[l] = v;
        }
    }
    __syncthreads();
    if (tid == 0) {
        float mx = -3.4e38f;
        for (int l = 0; l < LX; ++l) mx = fmaxf(mx, p[l]);
        float sum = 0.0f;
        for (int l = 0; l < LX; ++l) { float e = __expf(p[l] - mx); p[l] = e; sum += e; }
        float inv = 1.0f / sum;
        for (int l = 0; l < LX; ++l) p[l] *= inv;
    }
    __syncthreads();
    for (int j = tid; j < 2048; j += 256) {
        float acc = 0.0f;
        const float* xe = x_enc + (size_t)b * LX * 2048 + j;
#pragma unroll 4
        for (int l = 0; l < LX; ++l) acc += p[l] * xe[(size_t)l * 2048];
        __bf16 cb = f2bf(acc);
        xfh[(size_t)b * 3072 + j]        = cb;    // input_feed for step t+1
        xh[(size_t)b * 3072 + 1024 + j]  = cb;    // ctx for pre-readout GEMM
    }
}

// ---------------------------------------------------------------------------
// Logits GEMM, fallback path: B staged f32->bf16 through registers.
// ---------------------------------------------------------------------------
__global__ __launch_bounds__(256) void k_logits(
    const __bf16* __restrict__ pre, const float* __restrict__ Wr,
    float* __restrict__ out)
{
    __shared__ __attribute__((aligned(16))) __bf16 lb[64 * 72]; // 64 rows pad 72
    int nBase = blockIdx.x * 64;
    int mBase = blockIdx.y * 128;
    int tid  = threadIdx.x;
    int lane = tid & 31, w = tid >> 5;
    int ln = lane & 15, hf = lane >> 4;

    v8f acc[4] = {};
    const __bf16* aRow = pre + (size_t)(mBase + w * 16 + ln) * DM + 8 * hf;

    for (int kc = 0; kc < DM; kc += 64) {
        __syncthreads();
        {   // stage 64x64 f32 -> bf16 into LDS
            int row = tid >> 2, seg = tid & 3;
            const float* src = Wr + (size_t)(nBase + row) * DM + kc + seg * 16;
            __bf16* dst = lb + row * 72 + seg * 16;
#pragma unroll
            for (int e = 0; e < 16; ++e) dst[e] = f2bf(src[e]);
        }
        __syncthreads();
        FragBF a0, a1;
        load_a_bf(aRow, kc, a0);
        load_a_bf(aRow, kc + 32, a1);
#pragma unroll
        for (int ks = 0; ks < 64; ks += 32) {
            FragBF bfr[4];
#pragma unroll
            for (int nt = 0; nt < 4; ++nt) {
                const __bf16* bp = lb + (nt * 16 + ln) * 72 + ks + 8 * hf;
                bfr[nt].q[0] = *(const u32x4*)(bp);
                bfr[nt].q[1] = *(const u32x4*)(bp + 16);
            }
            wmma4(ks ? a1 : a0, bfr, acc);
        }
    }

#pragma unroll
    for (int nt = 0; nt < 4; ++nt) {
        int col = nBase + nt * 16 + ln;
#pragma unroll
        for (int r = 0; r < 8; ++r) {
            int row = mBase + w * 16 + 8 * hf + r;   // row = t*32 + b
            int t = row >> 5, b = row & 31;
            out[((size_t)b * TT + t) * VV + col] = acc[nt][r];
        }
    }
}

// ---------------------------------------------------------------------------
// Logits GEMM, preferred path: bf16 readout image + double-buffered
// global_load_async_to_lds_b128 staging (ASYNCcnt), WMMA from LDS.
// Each thread copies 2x16B per 64-wide K chunk (64 rows x 128B tile).
// ---------------------------------------------------------------------------
__global__ __launch_bounds__(256) void k_logits_async(
    const __bf16* __restrict__ pre, const __bf16* __restrict__ Wrb,
    float* __restrict__ out)
{
    __shared__ __attribute__((aligned(16))) __bf16 lb[2][64 * 72];
    int nBase = blockIdx.x * 64;
    int mBase = blockIdx.y * 128;
    int tid  = threadIdx.x;
    int lane = tid & 31, w = tid >> 5;
    int ln = lane & 15, hf = lane >> 4;

    v8f acc[4] = {};
    const __bf16* aRow = pre + (size_t)(mBase + w * 16 + ln) * DM + 8 * hf;

    // This thread's two 16-byte chunks of the 64x64 bf16 B tile.
    int row0 = tid >> 3,         seg = tid & 7;          // chunk 0
    int row1 = (tid + 256) >> 3;                          // chunk 1 (same seg)
    const __bf16* g0 = Wrb + (size_t)(nBase + row0) * DM + seg * 8;
    const __bf16* g1 = Wrb + (size_t)(nBase + row1) * DM + seg * 8;
    __bf16* l0[2] = { &lb[0][row0 * 72 + seg * 8], &lb[1][row0 * 72 + seg * 8] };
    __bf16* l1[2] = { &lb[0][row1 * 72 + seg * 8], &lb[1][row1 * 72 + seg * 8] };

    async_b128(l0[0], g0);
    async_b128(l1[0], g1);
    for (int kc = 0; kc < DM; kc += 64) {
        int buf = (kc >> 6) & 1;
        if (kc + 64 < DM) {
            async_b128(l0[buf ^ 1], g0 + kc + 64);
            async_b128(l1[buf ^ 1], g1 + kc + 64);
            asm volatile("s_wait_asynccnt 0x2" ::: "memory");  // current buf done
        } else {
            asm volatile("s_wait_asynccnt 0x0" ::: "memory");
        }
        __syncthreads();
        const __bf16* lbb = lb[buf];
        FragBF a0, a1;
        load_a_bf(aRow, kc, a0);
        load_a_bf(aRow, kc + 32, a1);
#pragma unroll
        for (int ks = 0; ks < 64; ks += 32) {
            FragBF bfr[4];
#pragma unroll
            for (int nt = 0; nt < 4; ++nt) {
                const __bf16* bp = lbb + (nt * 16 + ln) * 72 + ks + 8 * hf;
                bfr[nt].q[0] = *(const u32x4*)(bp);
                bfr[nt].q[1] = *(const u32x4*)(bp + 16);
            }
            wmma4(ks ? a1 : a0, bfr, acc);
        }
        __syncthreads();   // reads done before buf is re-staged next iter
    }

#pragma unroll
    for (int nt = 0; nt < 4; ++nt) {
        int col = nBase + nt * 16 + ln;
#pragma unroll
        for (int r = 0; r < 8; ++r) {
            int row = mBase + w * 16 + 8 * hf + r;   // row = t*32 + b
            int t = row >> 5, b = row & 31;
            out[((size_t)b * TT + t) * VV + col] = acc[nt][r];
        }
    }
}

// ---------------------------------------------------------------------------
// Host launcher
// ---------------------------------------------------------------------------
extern "C" void kernel_launch(void* const* d_in, const int* in_sizes, int n_in,
                              void* d_out, int out_size, void* d_ws, size_t ws_size,
                              hipStream_t stream) {
    const float* x_enc    = (const float*)d_in[0];
    const float* x_enc_k  = (const float*)d_in[1];
    const float* h0       = (const float*)d_in[2];
    const float* c0       = (const float*)d_in[3];
    const unsigned char* x_mask = (const unsigned char*)d_in[4];
    const int*   y_train  = (const int*)d_in[5];
    const float* word_emb = (const float*)d_in[6];
    const float* W_ih     = (const float*)d_in[7];
    const float* W_hh     = (const float*)d_in[8];
    const float* b_ih     = (const float*)d_in[9];
    const float* b_hh     = (const float*)d_in[10];
    const float* w_trg_W  = (const float*)d_in[11];
    const float* w_trg_b  = (const float*)d_in[12];
    const float* w_att_W  = (const float*)d_in[13];
    const float* w_att_b  = (const float*)d_in[14];
    const float* c2r_W    = (const float*)d_in[15];
    const float* readout_W= (const float*)d_in[16];
    float* out = (float*)d_out;

    char* ws = (char*)d_ws;
    size_t off = 0;
    auto alloc = [&](size_t bytes) { size_t p = off; off += (bytes + 255) & ~(size_t)255; return p; };

    __bf16* wfull  = (__bf16*)(ws + alloc((size_t)4096 * 4096 * 2)); // 33.5MB (L2-resident)
    __bf16* wtrg   = (__bf16*)(ws + alloc((size_t)1024 * 1024 * 2));
    __bf16* c2r    = (__bf16*)(ws + alloc((size_t)1024 * 3072 * 2));
    float*  bsum   = (float*) (ws + alloc(4096 * 4));
    __bf16* xfh    = (__bf16*)(ws + alloc((size_t)BD * 3072 * 2));   // [feed|h]
    __bf16* xh     = (__bf16*)(ws + alloc((size_t)BD * 3072 * 2));   // [h|ctx]
    float*  gates  = (float*) (ws + alloc((size_t)BD * 4096 * 4));
    float*  c_st   = (float*) (ws + alloc((size_t)BD * DM * 4));
    float*  tmpatt = (float*) (ws + alloc((size_t)BD * DM * 4));
    __bf16* preall = (__bf16*)(ws + alloc((size_t)TT * BD * DM * 2));
    size_t  base_need = off;
    __bf16* rbf    = (__bf16*)(ws + alloc((size_t)VV * DM * 2));     // 65.5MB optional
    bool use_async = (ws_size >= off) && (base_need <= ws_size);

    // --- prep: bf16 weight images, fused bias, initial state ---
    k_prep_wfull<<<(4096 * 4096) / 256, 256, 0, stream>>>(W_ih, W_hh, wfull);
    k_f2bf<<<(1024 * 1024) / 256, 256, 0, stream>>>(w_trg_W, wtrg, 1024 * 1024);
    k_f2bf<<<(1024 * 3072) / 256, 256, 0, stream>>>(c2r_W, c2r, 1024 * 3072);
    k_bsum<<<4096 / 256, 256, 0, stream>>>(b_ih, b_hh, bsum, 4096);
    k_init<<<(BD * 2048) / 256, 256, 0, stream>>>(h0, c0, c_st, xfh, xh);
    if (use_async)
        k_f2bf<<<(VV * DM) / 256, 256, 0, stream>>>(readout_W, rbf, VV * DM);

    // --- sequential decode: 64 dependent steps, 5 launches each ---
    for (int t = 0; t < TT; ++t) {
        k_gates<<<16, 256, 0, stream>>>(word_emb, y_train, t, xfh, wfull, bsum, gates);
        k_lstm<<<(BD * DM) / 256, 256, 0, stream>>>(gates, c_st, xfh, xh);
        // tmp = h @ w_trg_W^T + w_trg_b   (A = h region of xfh, lda 3072)
        k_gemm32<<<4, 256, 0, stream>>>(xfh + 2048, 3072, wtrg, 1024, w_trg_b,
                                        tmpatt, nullptr, 1024, 1024, 0);
        k_att<<<BD, 256, 0, stream>>>(x_enc_k, x_enc, tmpatt, w_att_W, w_att_b,
                                      x_mask, xfh, xh);
        // pre = tanh([h|ctx] @ ctx_to_readout^T)  -> bf16 rows of final A
        k_gemm32<<<4, 256, 0, stream>>>(xh, 3072, c2r, 3072, nullptr,
                                        nullptr, preall + (size_t)t * BD * DM,
                                        1024, 3072, 1);
    }

    // --- logits: [2048,1024] x [32000,1024]^T, bf16 WMMA ---
    dim3 grid(VV / 64, (TT * BD) / 128);
    if (use_async)
        k_logits_async<<<grid, 256, 0, stream>>>(preall, rbf, out);
    else
        k_logits<<<grid, 256, 0, stream>>>(preall, readout_W, out);
}
